// AttentiveTransformer_35476429865457
// MI455X (gfx1250) — matrix operand order repaired and checked by reading, not compiled, not run on hardware
//
#include <hip/hip_runtime.h>
#include <hip/hip_bf16.h>

// ---------------------------------------------------------------------------
// AttentiveTransformer: y = feat @ W^T ; GhostBatchNorm(VBS=256) ; * priors ;
// sparsemax(rows).  MI455X (gfx1250, wave32, WMMA, async-to-LDS).
//
// Pipeline:
//   1) bf16_split_kernel: one-time f32 -> (bf16 hi, bf16 lo) for A and W.
//   2) gemm_bf16split_kernel: 3-WMMA split GEMM (hi*hi + hi*lo + lo*hi),
//      double-buffered LDS filled by GLOBAL_LOAD_ASYNC_TO_LDS_B128 so the
//      next K-slab streams in while WMMAs run on the current one.
//   3) gbn_stats_kernel: per-(virtual batch, column) mean/rstd.
//   4) bn_sparsemax_kernel: BN-apply * priors + bisection sparsemax per row.
// ---------------------------------------------------------------------------

typedef __attribute__((ext_vector_type(16))) __bf16 v16bf;
typedef __attribute__((ext_vector_type(8)))  __bf16 v8bf;
typedef __attribute__((ext_vector_type(4)))  __bf16 v4bf;
typedef __attribute__((ext_vector_type(8)))  float  v8f;

#define TILE_M 128
#define TILE_N 128
#define TILE_K 32
#define LROW   40    // bf16 row stride (80B) for LDS bank spread, keeps 16B align
#define VBS    256

// --- CDNA5 async global->LDS (ASYNCcnt-tracked), inline asm path -----------
__device__ __forceinline__ void async_load16B(const void* gaddr, const void* lds) {
    // VDST = LDS byte address (low 32 bits of flat LDS-aperture address),
    // VADDR = 64-bit global address, GV mode (SADDR = off).
    asm volatile("global_load_async_to_lds_b128 %0, %1, off"
                 :: "v"((unsigned)(uintptr_t)lds), "v"(gaddr)
                 : "memory");
}
#define WAIT_ASYNC(n) asm volatile("s_wait_asynccnt %0" :: "n"(n) : "memory")

// ---------------------------------------------------------------------------
// Kernel 0: split f32 into bf16 hi + bf16 lo (x ~= hi + lo), 4 elems/thread.
// ---------------------------------------------------------------------------
__global__ __launch_bounds__(256) void bf16_split_kernel(
    const float* __restrict__ X, __bf16* __restrict__ Hi,
    __bf16* __restrict__ Lo, int n4)
{
    const int i = blockIdx.x * blockDim.x + threadIdx.x;
    if (i >= n4) return;
    const float4 v = ((const float4*)X)[i];
    v4bf h, l;
    float f;
    f = v.x; h[0] = (__bf16)f; l[0] = (__bf16)(f - (float)h[0]);
    f = v.y; h[1] = (__bf16)f; l[1] = (__bf16)(f - (float)h[1]);
    f = v.z; h[2] = (__bf16)f; l[2] = (__bf16)(f - (float)h[2]);
    f = v.w; h[3] = (__bf16)f; l[3] = (__bf16)(f - (float)h[3]);
    ((v4bf*)Hi)[i] = h;
    ((v4bf*)Lo)[i] = l;
}

// concat two aligned 8x bf16 LDS vectors into one 16x fragment (register-only)
__device__ __forceinline__ v16bf frag16(const __bf16* p0, const __bf16* p1) {
    const v8bf x = *(const v8bf*)p0;
    const v8bf y = *(const v8bf*)p1;
    return __builtin_shufflevector(x, y, 0, 1, 2, 3, 4, 5, 6, 7,
                                         8, 9, 10, 11, 12, 13, 14, 15);
}

// ---------------------------------------------------------------------------
// Kernel 1: Y[M,N] = A[M,K] @ W[N,K]^T  using pre-split bf16 hi/lo operands.
// Block: 256 threads = 8 waves.  Block tile 128x128, wave tile 16x128.
// Double-buffered async staging: tile k+1 streams in during WMMAs on tile k.
// ---------------------------------------------------------------------------
__global__ __launch_bounds__(256) void gemm_bf16split_kernel(
    const __bf16* __restrict__ Ahi, const __bf16* __restrict__ Alo,
    const __bf16* __restrict__ Whi, const __bf16* __restrict__ Wlo,
    float* __restrict__ Y, int N, int K)
{
    __shared__ __bf16 sAhi[2][TILE_M][LROW];
    __shared__ __bf16 sAlo[2][TILE_M][LROW];
    __shared__ __bf16 sBhi[2][TILE_N][LROW];
    __shared__ __bf16 sBlo[2][TILE_N][LROW];

    const int tid  = threadIdx.x;
    const int wave = tid >> 5;
    const int lane = tid & 31;
    const int m0   = blockIdx.y * TILE_M;
    const int n0   = blockIdx.x * TILE_N;

    v8f acc[8];
#pragma unroll
    for (int i = 0; i < 8; ++i)
#pragma unroll
        for (int j = 0; j < 8; ++j) acc[i][j] = 0.0f;

    // fragment lane layouts (ISA 16-bit A 16x32 / B 32x16)
    const int arow = wave * 16 + (lane & 15);
    const int akb  = (lane >> 4) * 8;   // lanes<16: K 0-7,16-23 ; lanes>=16: K 8-15,24-31
    const int bcol = lane & 15;
    const int bkb  = (lane >> 4) * 8;   // lanes<16: K 0-7,16-23 ; lanes>=16: K 8-15,24-31

    // staging assignment: thread -> (row, 16-elem half of the 32-wide K slab)
    const int sr = tid >> 1;            // 0..127
    const int sh = (tid & 1) * 16;      // 0 or 16

    // issue one tile's worth of async loads (8 x 16B per thread) into buffer b
    auto issue = [&](int b, int k0) {
        const size_t ga = (size_t)(m0 + sr) * K + k0 + sh;
        const size_t gb = (size_t)(n0 + sr) * K + k0 + sh;
        async_load16B(Ahi + ga,     &sAhi[b][sr][sh]);
        async_load16B(Ahi + ga + 8, &sAhi[b][sr][sh + 8]);
        async_load16B(Alo + ga,     &sAlo[b][sr][sh]);
        async_load16B(Alo + ga + 8, &sAlo[b][sr][sh + 8]);
        async_load16B(Whi + gb,     &sBhi[b][sr][sh]);
        async_load16B(Whi + gb + 8, &sBhi[b][sr][sh + 8]);
        async_load16B(Wlo + gb,     &sBlo[b][sr][sh]);
        async_load16B(Wlo + gb + 8, &sBlo[b][sr][sh + 8]);
    };

    const int nK = K / TILE_K;
    issue(0, 0);

    for (int t = 0; t < nK; ++t) {
        const int cur = t & 1;
        if (t + 1 < nK) {
            issue(cur ^ 1, (t + 1) * TILE_K);
            WAIT_ASYNC(8);   // async loads complete in order: first 8 done
        } else {
            WAIT_ASYNC(0);
        }
        __syncthreads();     // current buffer fully resident for all waves

        // this wave's A fragments: two aligned 16B LDS loads each
        const v16bf a_hi = frag16(&sAhi[cur][arow][akb], &sAhi[cur][arow][16 + akb]);
        const v16bf a_lo = frag16(&sAlo[cur][arow][akb], &sAlo[cur][arow][16 + akb]);

#pragma unroll
        for (int sub = 0; sub < 8; ++sub) {
            const int brow = sub * 16 + bcol;
            const v16bf b_hi = frag16(&sBhi[cur][brow][bkb], &sBhi[cur][brow][bkb + 16]);
            const v16bf b_lo = frag16(&sBlo[cur][brow][bkb], &sBlo[cur][brow][bkb + 16]);

            // smallest terms first for accumulation accuracy
            acc[sub] = __builtin_amdgcn_wmma_f32_16x16x32_bf16(
                false, a_lo, false, b_hi, (short)0, acc[sub], false, false);
            acc[sub] = __builtin_amdgcn_wmma_f32_16x16x32_bf16(
                false, a_hi, false, b_lo, (short)0, acc[sub], false, false);
            acc[sub] = __builtin_amdgcn_wmma_f32_16x16x32_bf16(
                false, a_hi, false, b_hi, (short)0, acc[sub], false, false);
        }
        __syncthreads();     // all waves done reading before buffer is re-filled
    }

    // epilogue: C/D layout — VGPR r: M = r + (lane>=16 ? 8:0), N = lane&15
    const int nlane = lane & 15;
    const int mhalf = (lane >> 4) * 8;
#pragma unroll
    for (int sub = 0; sub < 8; ++sub) {
#pragma unroll
        for (int r = 0; r < 8; ++r) {
            const int m = m0 + wave * 16 + mhalf + r;
            const int n = n0 + sub * 16 + nlane;
            Y[(size_t)m * N + n] = acc[sub][r];
        }
    }
}

// ---------------------------------------------------------------------------
// Kernel 2: GhostBatchNorm stats — one thread per (virtual batch, column).
// ---------------------------------------------------------------------------
__global__ __launch_bounds__(256) void gbn_stats_kernel(
    const float* __restrict__ Y, float* __restrict__ mean,
    float* __restrict__ rstd, int N)
{
    const int col = blockIdx.x * blockDim.x + threadIdx.x;
    const int vb  = blockIdx.y;
    const float* p = Y + (size_t)vb * VBS * N + col;
    float s = 0.f, ss = 0.f;
#pragma unroll 4
    for (int r = 0; r < VBS; ++r) {
        const float v = p[(size_t)r * N];
        s += v; ss += v * v;
    }
    const float mu  = s * (1.0f / VBS);
    const float var = ss * (1.0f / VBS) - mu * mu;
    mean[(size_t)vb * N + col] = mu;
    rstd[(size_t)vb * N + col] = rsqrtf(var + 1e-5f);
}

// ---------------------------------------------------------------------------
// Kernel 3: per-row BN apply * priors, then sparsemax via bisection on tau.
// ---------------------------------------------------------------------------
__device__ __forceinline__ float block_sum(float v, float* sm8) {
#pragma unroll
    for (int off = 16; off > 0; off >>= 1) v += __shfl_xor(v, off, 32);
    if ((threadIdx.x & 31) == 0) sm8[threadIdx.x >> 5] = v;
    __syncthreads();
    float t = 0.f;
#pragma unroll
    for (int w = 0; w < 8; ++w) t += sm8[w];
    __syncthreads();
    return t;
}

__device__ __forceinline__ float block_max(float v, float* sm8) {
#pragma unroll
    for (int off = 16; off > 0; off >>= 1) v = fmaxf(v, __shfl_xor(v, off, 32));
    if ((threadIdx.x & 31) == 0) sm8[threadIdx.x >> 5] = v;
    __syncthreads();
    float t = -3.4e38f;
#pragma unroll
    for (int w = 0; w < 8; ++w) t = fmaxf(t, sm8[w]);
    __syncthreads();
    return t;
}

#define ITEMS 8  // N / 256

__global__ __launch_bounds__(256) void bn_sparsemax_kernel(
    const float* __restrict__ Y, const float* __restrict__ priors,
    const float* __restrict__ gamma, const float* __restrict__ beta,
    const float* __restrict__ mean, const float* __restrict__ rstd,
    float* __restrict__ out, int N)
{
    __shared__ float sm8[8];
    const int row = blockIdx.x;
    const int vb  = row >> 8;           // row / VBS
    const int tid = threadIdx.x;

    float z[ITEMS];
#pragma unroll
    for (int i = 0; i < ITEMS; ++i) {
        const int col = tid + i * 256;
        const size_t rc = (size_t)row * N + col;
        const size_t vc = (size_t)vb * N + col;
        const float bn = (Y[rc] - mean[vc]) * rstd[vc] * gamma[col] + beta[col];
        z[i] = bn * priors[rc];
    }

    float m = z[0];
#pragma unroll
    for (int i = 1; i < ITEMS; ++i) m = fmaxf(m, z[i]);
    const float rowmax = block_max(m, sm8);
#pragma unroll
    for (int i = 0; i < ITEMS; ++i) z[i] -= rowmax;

    // f(tau) = sum(max(z - tau, 0)) monotone decreasing; f(-1) >= 1, f(0) < 1
    float lo = -1.0f, hi = 0.0f;
    for (int it = 0; it < 32; ++it) {
        const float tau = 0.5f * (lo + hi);
        float s = 0.f;
#pragma unroll
        for (int i = 0; i < ITEMS; ++i) s += fmaxf(z[i] - tau, 0.f);
        const float tot = block_sum(s, sm8);
        if (tot >= 1.0f) lo = tau; else hi = tau;
    }
    const float tau = 0.5f * (lo + hi);

#pragma unroll
    for (int i = 0; i < ITEMS; ++i) {
        const int col = tid + i * 256;
        out[(size_t)row * N + col] = fmaxf(z[i] - tau, 0.f);
    }
}

// ---------------------------------------------------------------------------
// Launch.  Inputs: priors[B,N], processed_feat[B,K], W[N,K], gamma[N], beta[N]
// ---------------------------------------------------------------------------
extern "C" void kernel_launch(void* const* d_in, const int* in_sizes, int n_in,
                              void* d_out, int out_size, void* d_ws, size_t ws_size,
                              hipStream_t stream) {
    const float* priors = (const float*)d_in[0];
    const float* feat   = (const float*)d_in[1];
    const float* W      = (const float*)d_in[2];
    const float* gamma  = (const float*)d_in[3];
    const float* beta   = (const float*)d_in[4];

    const int N = in_sizes[3];                 // D_out = 2048
    const int K = in_sizes[2] / N;             // D_in  = 512
    const int B = in_sizes[1] / K;             // batch = 32768
    const int NVB = B / VBS;                   // 128 virtual batches

    // workspace layout
    char* ws = (char*)d_ws;
    float* y    = (float*)ws;                               // [B, N] f32
    float* mean = (float*)(ws + (size_t)B * N * 4);         // [NVB, N]
    float* rstd = mean + (size_t)NVB * N;                   // [NVB, N]
    char* p = (char*)(rstd + (size_t)NVB * N);
    __bf16* Ahi = (__bf16*)p; p += (size_t)B * K * 2;
    __bf16* Alo = (__bf16*)p; p += (size_t)B * K * 2;
    __bf16* Whi = (__bf16*)p; p += (size_t)N * K * 2;
    __bf16* Wlo = (__bf16*)p;
    float* outp = (float*)d_out;

    // 0) one-time f32 -> bf16 hi/lo split of both GEMM operands
    const int nA4 = (B * K) / 4;
    const int nW4 = (N * K) / 4;
    bf16_split_kernel<<<(nA4 + 255) / 256, 256, 0, stream>>>(feat, Ahi, Alo, nA4);
    bf16_split_kernel<<<(nW4 + 255) / 256, 256, 0, stream>>>(W, Whi, Wlo, nW4);

    // 1) WMMA GEMM (async double-buffered staging)
    dim3 gGemm(N / TILE_N, B / TILE_M);
    gemm_bf16split_kernel<<<gGemm, 256, 0, stream>>>(Ahi, Alo, Whi, Wlo, y, N, K);

    // 2) GhostBatchNorm stats
    dim3 gStats(N / 256, NVB);
    gbn_stats_kernel<<<gStats, 256, 0, stream>>>(y, mean, rstd, N);

    // 3) BN apply * priors + sparsemax
    bn_sparsemax_kernel<<<B, 256, 0, stream>>>(y, priors, gamma, beta,
                                               mean, rstd, outp, N);
}
